// FuturePredictionODE_3186865734367
// MI455X (gfx1250) — matrix operand order repaired and checked by reading, not compiled, not run on hardware
//
#include <hip/hip_runtime.h>
#include <math.h>

// ---------------------------------------------------------------------------
// FuturePredictionODE on gfx1250: implicit-GEMM convs via v_wmma_f32_16x16x32_bf16
// All conv shapes are compile-time (templated) -> no runtime int division.
// One block = 16 pixels x ALL output channels (NT n-tiles per wave, 4 waves).
// ---------------------------------------------------------------------------

#define DT_F 0.05f

typedef __bf16 bf16_t;
typedef bf16_t v16bf __attribute__((ext_vector_type(16)));
typedef float v8f __attribute__((ext_vector_type(8)));
typedef float v4f __attribute__((ext_vector_type(4)));
typedef unsigned short v16us __attribute__((ext_vector_type(16)));

static constexpr int B_ = 2, C_ = 64, H_ = 64, W_ = 64;
static constexpr int HW_ = H_ * W_;
static constexpr int NC_ = 3, NL_ = 5, NFR_ = 6;
static constexpr int NSTEP_ = NC_ + NL_ + NFR_;          // 14
static constexpr long FRAME_ = (long)C_ * HW_;           // elems per (b) frame @64ch

__device__ __forceinline__ unsigned short f2bf(float f) {
  unsigned int u = __float_as_uint(f);
  u += 0x7FFFu + ((u >> 16) & 1u);   // round-to-nearest-even
  return (unsigned short)(u >> 16);
}
__device__ __forceinline__ float sigm(float x) { return 1.0f / (1.0f + __expf(-x)); }

// ---------------------------------------------------------------------------
// Weight pack: OIHW f32 -> bf16 in exact B-fragment order:
//   packed[ntile][kc][lane][e], lane&15 = N col, (lane>>4)*16 .. = K half
// ---------------------------------------------------------------------------
template <int CIN, int KHW>
__global__ void k_pack_w(const float* __restrict__ w, unsigned short* __restrict__ out,
                         int Cout, int n) {
  constexpr int KC = (CIN * KHW) / 32;
  int tid = blockIdx.x * blockDim.x + threadIdx.x;
  if (tid >= n) return;
  int e    = tid & 15;
  int lane = (tid >> 4) & 31;
  int rest = tid >> 9;
  int kc    = rest % KC;
  int ntile = rest / KC;
  int col   = lane & 15;
  int kbase = (lane >> 4) * 16;
  int k  = kc * 32 + kbase + e;
  int co = ntile * 16 + col;
  int ci = k / KHW;                    // constant divisor
  int r  = k - ci * KHW;
  float v = 0.0f;
  if (co < Cout && ci < CIN) v = w[((long)co * CIN + ci) * KHW + r];
  out[tid] = f2bf(v);
}

// ---------------------------------------------------------------------------
// Implicit-GEMM conv, bf16 WMMA, f32 accumulate.
// Block = 128 threads (4 waves). Tile: 16 pixels (M) x 64*NT couts.
// Wave w handles n-tiles {w, w+4, ..., w+4*(NT-1)}. gridDim.x = Nimg*H*(W/16).
// ---------------------------------------------------------------------------
template <int CIN, int KH, int NT>
__global__ __launch_bounds__(128)
void k_conv_wmma(const float* __restrict__ X, const unsigned short* __restrict__ Wp,
                 float* __restrict__ Y, long inStride, long outStride) {
  constexpr int KHW = KH * KH;
  constexpr int KC  = (CIN * KHW) / 32;   // exact for all used shapes
  constexpr int pad = KH / 2;
  static_assert(W_ == 64 && H_ == 64, "decode assumes 64x64");

  int bx = blockIdx.x;
  int wbase = (bx & 3) << 4;              // W/16 == 4 tiles per row
  int hh = (bx >> 2) & (H_ - 1);
  int n  = bx >> 8;                       // / (H*(W/16)) == 256

  int tid  = threadIdx.x;
  int lane = tid & 31;
  int wave = tid >> 5;

  // A-fragment staging in LDS, already in WMMA A lane layout (32 lanes x 16 bf16).
  __shared__ __attribute__((aligned(32))) unsigned short sA[32 * 16];

  // This thread builds 4 consecutive elements of fragment-lane l.
  int l  = tid >> 2;
  int i0 = (tid & 3) << 2;
  int arow  = l & 15;                     // M within tile (pixel)
  int khalf = l >> 4;                     // K half select (A layout)
  int wcol  = wbase + arow;
  const float* Xn = X + (long)n * inStride;

  // k-offsets (within a 32-chunk) of this thread's 4 elements.
  int koff[4];
#pragma unroll
  for (int e = 0; e < 4; ++e) {
    int i = i0 + e;
    koff[e] = khalf * 16 + i;
  }

  const unsigned short* wp0 = Wp + (long)lane * 16;   // + (ntile*KC+kc)*512

  v8f acc[NT];
#pragma unroll
  for (int t = 0; t < NT; ++t) acc[t] = (v8f){};

#pragma unroll
  for (int kc = 0; kc < KC; ++kc) {
    // ---- build 4 bf16 A elements, pack to one 8-byte LDS store ----
    unsigned short q[4];
#pragma unroll
    for (int e = 0; e < 4; ++e) {
      int k  = kc * 32 + koff[e];
      int ci = k / KHW;                   // constant divisors -> magic mults
      int rr = k - ci * KHW;
      int kh = rr / KH;
      int kw = rr - kh * KH;
      int hs = hh + kh - pad;
      int wsx = wcol + kw - pad;
      float v = 0.0f;
      if (hs >= 0 && hs < H_ && wsx >= 0 && wsx < W_)
        v = Xn[((long)ci * H_ + hs) * W_ + wsx];
      q[e] = f2bf(v);
    }
    {
      unsigned int lo = (unsigned int)q[0] | ((unsigned int)q[1] << 16);
      unsigned int hi = (unsigned int)q[2] | ((unsigned int)q[3] << 16);
      unsigned int* dst = (unsigned int*)(sA + l * 16 + i0);
      dst[0] = lo;
      dst[1] = hi;
    }
    __syncthreads();

    v16us au = *(const v16us*)(sA + lane * 16);
    v16bf a = __builtin_bit_cast(v16bf, au);
#pragma unroll
    for (int t = 0; t < NT; ++t) {
      int ntile = wave + 4 * t;
      const unsigned short* bp = wp0 + ((long)ntile * KC + kc) * 512;
      if (kc + 1 < KC) __builtin_prefetch(bp + 512, 0, 0);   // global_prefetch_b8
      v16us bu = *(const v16us*)bp;
      v16bf b = __builtin_bit_cast(v16bf, bu);
      acc[t] = __builtin_amdgcn_wmma_f32_16x16x32_bf16(false, a, false, b, (short)0,
                                                       acc[t], false, false);
    }
    __syncthreads();
  }

  // C/D layout: lane&15 = N col, lane>>4 picks M 0-7 / 8-15 across the 8 regs.
  int col  = lane & 15;
  int half = lane >> 4;
#pragma unroll
  for (int t = 0; t < NT; ++t) {
    int co = (wave + 4 * t) * 16 + col;
    float* Yn = Y + (long)n * outStride + ((long)co * H_ + hh) * W_ + wbase + half * 8;
    v4f lo = {acc[t][0], acc[t][1], acc[t][2], acc[t][3]};
    v4f hi = {acc[t][4], acc[t][5], acc[t][6], acc[t][7]};
    *(v4f*)(Yn + 0) = lo;
    *(v4f*)(Yn + 4) = hi;
  }
}

// ---------------------------------------------------------------------------
// Elementwise / schedule kernels
// ---------------------------------------------------------------------------
__global__ void k_copy2(float* __restrict__ dst, const float* __restrict__ src,
                        long srcBStride, int perB, int n) {
  int i = blockIdx.x * blockDim.x + threadIdx.x;
  if (i >= n) return;
  int b = i / perB;
  dst[i] = src[(long)b * srcBStride + (i - b * perB)];
}
__global__ void k_relu(float* x, int n) {
  int i = blockIdx.x * blockDim.x + threadIdx.x;
  if (i < n) x[i] = fmaxf(x[i], 0.0f);
}
__global__ void k_add(float* x, const float* y, int n) {
  int i = blockIdx.x * blockDim.x + threadIdx.x;
  if (i < n) x[i] += y[i];
}
// GRU gate fuse: h' = (1-z)h + z g; optional per-batch obs mask; optional seq out
__global__ void k_gru(float* __restrict__ h, const float* __restrict__ gx,
                      const float* __restrict__ gh, const int* __restrict__ msk,
                      float* __restrict__ y, long yStride, int n) {
  int i = blockIdx.x * blockDim.x + threadIdx.x;
  if (i >= n) return;
  int b = i / (C_ * HW_);
  int rem = i - b * (C_ * HW_);
  long g0 = (long)b * 3 * C_ * HW_ + rem;
  float hv = h[i];
  float r = sigm(gx[g0] + gh[g0]);
  float z = sigm(gx[g0 + (long)C_ * HW_] + gh[g0 + (long)C_ * HW_]);
  float g = tanhf(gx[g0 + 2L * C_ * HW_] + r * gh[g0 + 2L * C_ * HW_]);
  float hn = (1.0f - z) * hv + z * g;
  if (msk && msk[b] < 0) hn = hv;
  h[i] = hn;
  if (y) y[(long)b * yStride + rem] = hn;
}
// rh = sigmoid(rz[:, 0:C]) * h
__global__ void k_rh(float* __restrict__ rh, const float* __restrict__ rz,
                     const float* __restrict__ h, int n) {
  int i = blockIdx.x * blockDim.x + threadIdx.x;
  if (i >= n) return;
  int b = i / (C_ * HW_);
  int rem = i - b * (C_ * HW_);
  rh[i] = sigm(rz[(long)b * 2 * C_ * HW_ + rem]) * h[i];
}
// h += dt*(1-z)*(tanh(gconv)-h), z = sigmoid(rz[:, C:2C])
__global__ void k_ode(float* __restrict__ h, const float* __restrict__ rz,
                      const float* __restrict__ g, int n) {
  int i = blockIdx.x * blockDim.x + threadIdx.x;
  if (i >= n) return;
  int b = i / (C_ * HW_);
  int rem = i - b * (C_ * HW_);
  float z = sigm(rz[(long)b * 2 * C_ * HW_ + (long)C_ * HW_ + rem]);
  float hv = h[i];
  h[i] = hv + DT_F * (1.0f - z) * (tanhf(g[i]) - hv);
}
// xobs[b] = observation scheduled at this step (or zeros)
__global__ void k_gather_obs(float* __restrict__ xo, const float* __restrict__ cam,
                             const float* __restrict__ lid, const int* __restrict__ slot,
                             int n) {
  int i = blockIdx.x * blockDim.x + threadIdx.x;
  if (i >= n) return;
  int b = i / (C_ * HW_);
  int rem = i - b * (C_ * HW_);
  int j = slot[b];
  float v = 0.0f;
  if (j >= 0) {
    if (j < NC_) v = cam[((long)b * NC_ + j) * C_ * HW_ + rem];
    else         v = lid[((long)b * NL_ + (j - NC_)) * C_ * HW_ + rem];
  }
  xo[i] = v;
}
// capture h into xseq[b,f] when tsteps[b,f]==t
__global__ void k_capture(float* __restrict__ xseq, const float* __restrict__ h,
                          const int* __restrict__ tst, int t, int n) {
  int i = blockIdx.x * blockDim.x + threadIdx.x;
  if (i >= n) return;
  int bf = i / (C_ * HW_);
  if (tst[bf] != t) return;
  int b = bf / NFR_;
  xseq[i] = h[(long)b * C_ * HW_ + (i - bf * (C_ * HW_))];
}
// device-side schedule: slot[t*B+b] = obs index at step t (-1 none); tst = round(t/dt)-1
__global__ void k_sched(const float* camt, const float* lidt, const float* tgt,
                        int* slot, int* tst) {
  int b = threadIdx.x;
  if (b >= B_) return;
  for (int t = 0; t < NSTEP_; ++t) slot[t * B_ + b] = -1;
  for (int j = 0; j < NC_; ++j) {
    int s = (int)rintf(camt[b * NC_ + j] / DT_F);
    if (s >= 0 && s < NSTEP_) slot[s * B_ + b] = j;
  }
  for (int j = 0; j < NL_; ++j) {
    int s = (int)rintf(lidt[b * NL_ + j] / DT_F);
    if (s >= 0 && s < NSTEP_) slot[s * B_ + b] = NC_ + j;
  }
  for (int f = 0; f < NFR_; ++f)
    tst[b * NFR_ + f] = (int)rintf(tgt[b * NFR_ + f] / DT_F) - 1;
}
__global__ void k_aux(float* p) { *p = 0.0f; }

// ---------------------------------------------------------------------------
// Host orchestration
// ---------------------------------------------------------------------------
template <int CIN, int KH, int NT>
static void conv(hipStream_t s, const float* X, const unsigned short* Wp, float* Y,
                 int Nimg, long inStride, long outStride) {
  dim3 g(Nimg * H_ * (W_ / 16));
  k_conv_wmma<CIN, KH, NT><<<g, 128, 0, s>>>(X, Wp, Y, inStride, outStride);
}
template <int CIN, int KHW>
static void pack(hipStream_t s, const float* w, unsigned short* p, int Cout) {
  int n = Cout * CIN * KHW;
  k_pack_w<CIN, KHW><<<(n + 255) / 256, 256, 0, s>>>(w, p, Cout, n);
}
static inline int eb(int n) { return (n + 255) / 256; }

extern "C" void kernel_launch(void* const* d_in, const int* /*in_sizes*/, int /*n_in*/,
                              void* d_out, int /*out_size*/, void* d_ws, size_t /*ws_size*/,
                              hipStream_t stream) {
  const float* x0   = (const float*)d_in[0];
  const float* cam  = (const float*)d_in[1];
  const float* lid  = (const float*)d_in[2];
  const float* camt = (const float*)d_in[3];
  const float* lidt = (const float*)d_in[4];
  const float* tgt  = (const float*)d_in[5];
  const float* w_orz = (const float*)d_in[6];
  const float* w_og  = (const float*)d_in[7];
  const float* w_jx  = (const float*)d_in[8];
  const float* w_jh  = (const float*)d_in[9];
  const float* w_s0x = (const float*)d_in[10];
  const float* w_s0h = (const float*)d_in[11];
  const float* w_s1x = (const float*)d_in[12];
  const float* w_s1h = (const float*)d_in[13];
  const float* w_r1  = (const float*)d_in[14];
  const float* w_r2  = (const float*)d_in[15];
  const float* w_d1  = (const float*)d_in[16];
  const float* w_d2  = (const float*)d_in[17];

  char* ws = (char*)d_ws;
  size_t off = 0;
  auto take = [&](size_t bytes) -> void* {
    void* p = ws + off;
    off = (off + bytes + 255) & ~(size_t)255;
    return p;
  };
  const int nE   = B_ * C_ * HW_;                 // 524288
  const int nSeq = B_ * NFR_ * C_ * HW_;          // 3145728
  float* xobs   = (float*)take(sizeof(float) * nE);
  float* h      = (float*)take(sizeof(float) * nE);
  float* hidden = (float*)take(sizeof(float) * nE);
  float* gx     = (float*)take(sizeof(float) * (size_t)B_ * 3 * C_ * HW_);
  float* gh     = (float*)take(sizeof(float) * (size_t)B_ * 3 * C_ * HW_);
  float* rz     = (float*)take(sizeof(float) * (size_t)B_ * 2 * C_ * HW_);
  float* rh     = (float*)take(sizeof(float) * nE);
  float* xseq   = (float*)take(sizeof(float) * (size_t)nSeq);
  float* t128   = (float*)take(sizeof(float) * (size_t)B_ * NFR_ * 128 * HW_);
  float* t64b   = (float*)take(sizeof(float) * (size_t)nSeq);
  unsigned short* p_jx  = (unsigned short*)take(2uL * 192 * 576);
  unsigned short* p_jh  = (unsigned short*)take(2uL * 192 * 576);
  unsigned short* p_orz = (unsigned short*)take(2uL * 128 * 576);
  unsigned short* p_og  = (unsigned short*)take(2uL * 64 * 576);
  unsigned short* p_s0x = (unsigned short*)take(2uL * 192 * 576);
  unsigned short* p_s0h = (unsigned short*)take(2uL * 192 * 576);
  unsigned short* p_s1x = (unsigned short*)take(2uL * 192 * 576);
  unsigned short* p_s1h = (unsigned short*)take(2uL * 192 * 576);
  unsigned short* p_r1  = (unsigned short*)take(2uL * 64 * 576);
  unsigned short* p_r2  = (unsigned short*)take(2uL * 64 * 576);
  unsigned short* p_d1  = (unsigned short*)take(2uL * 128 * 576);
  unsigned short* p_d2  = (unsigned short*)take(2uL * 64 * 128);
  int* slot = (int*)take(sizeof(int) * NSTEP_ * B_);
  int* tst  = (int*)take(sizeof(int) * B_ * NFR_);

  // Weight packs (every call; no cross-call state allowed).
  pack<64, 9>(stream, w_jx,  p_jx,  192);
  pack<64, 9>(stream, w_jh,  p_jh,  192);
  pack<64, 9>(stream, w_orz, p_orz, 128);
  pack<64, 9>(stream, w_og,  p_og,   64);
  pack<64, 9>(stream, w_s0x, p_s0x, 192);
  pack<64, 9>(stream, w_s0h, p_s0h, 192);
  pack<64, 9>(stream, w_s1x, p_s1x, 192);
  pack<64, 9>(stream, w_s1h, p_s1h, 192);
  pack<64, 9>(stream, w_r1,  p_r1,   64);
  pack<64, 9>(stream, w_r2,  p_r2,   64);
  pack<64, 9>(stream, w_d1,  p_d1,  128);
  pack<128, 1>(stream, w_d2,  p_d2,  64);

  // Schedule from timestamp inputs (device-side -> deterministic, value-driven).
  k_sched<<<1, 32, 0, stream>>>(camt, lidt, tgt, slot, tst);

  // h = future_prediction_input
  k_copy2<<<eb(nE), 256, 0, stream>>>(h, x0, FRAME_, C_ * HW_, nE);

  const long s1  = FRAME_;            // contiguous 64ch frame stride
  const long s3  = 3 * FRAME_;
  const long s2  = 2 * FRAME_;
  const long sSq = (long)NFR_ * FRAME_;

  // ---- ODE scan over 14 steps: jump-GRU (masked) + ODE update + capture ----
  for (int t = 0; t < NSTEP_; ++t) {
    k_gather_obs<<<eb(nE), 256, 0, stream>>>(xobs, cam, lid, slot + t * B_, nE);
    conv<64, 3, 3>(stream, xobs, p_jx, gx, B_, s1, s3);
    conv<64, 3, 3>(stream, h,    p_jh, gh, B_, s1, s3);
    k_gru<<<eb(nE), 256, 0, stream>>>(h, gx, gh, slot + t * B_, nullptr, 0, nE);
    conv<64, 3, 2>(stream, h, p_orz, rz, B_, s1, s2);
    k_rh<<<eb(nE), 256, 0, stream>>>(rh, rz, h, nE);
    conv<64, 3, 1>(stream, rh, p_og, gx, B_, s1, s1);   // reuse gx as g-conv out
    k_ode<<<eb(nE), 256, 0, stream>>>(h, rz, gx, nE);
    k_capture<<<eb(nSeq), 256, 0, stream>>>(xseq, h, tst, t, nSeq);
  }

  // hidden = xseq[:, 0]
  k_copy2<<<eb(nE), 256, 0, stream>>>(hidden, xseq, sSq, C_ * HW_, nE);

  // ---- sequential GRU 0 ----
  k_copy2<<<eb(nE), 256, 0, stream>>>(h, hidden, s1, C_ * HW_, nE);
  for (int t = 0; t < NFR_; ++t) {
    conv<64, 3, 3>(stream, xseq + t * FRAME_, p_s0x, gx, B_, sSq, s3);
    conv<64, 3, 3>(stream, h,                 p_s0h, gh, B_, s1,  s3);
    k_gru<<<eb(nE), 256, 0, stream>>>(h, gx, gh, nullptr, xseq + t * FRAME_, sSq, nE);
  }

  // ---- residual block over B*NF frames ----
  const int nF12 = B_ * NFR_;
  conv<64, 3, 1>(stream, xseq, p_r1, t128, nF12, s1, s1);
  k_relu<<<eb(nSeq), 256, 0, stream>>>(t128, nSeq);
  conv<64, 3, 1>(stream, t128, p_r2, t64b, nF12, s1, s1);
  k_add<<<eb(nSeq), 256, 0, stream>>>(xseq, t64b, nSeq);

  // ---- sequential GRU 1 (same hidden) ----
  k_copy2<<<eb(nE), 256, 0, stream>>>(h, hidden, s1, C_ * HW_, nE);
  for (int t = 0; t < NFR_; ++t) {
    conv<64, 3, 3>(stream, xseq + t * FRAME_, p_s1x, gx, B_, sSq, s3);
    conv<64, 3, 3>(stream, h,                 p_s1h, gh, B_, s1,  s3);
    k_gru<<<eb(nE), 256, 0, stream>>>(h, gx, gh, nullptr, xseq + t * FRAME_, sSq, nE);
  }

  // ---- decode head: conv64->128 (3x3) + relu + conv128->64 (1x1) -> d_out ----
  float* outF = (float*)d_out;
  conv<64, 3, 2>(stream, xseq, p_d1, t128, nF12, s1, 2 * FRAME_);
  int n128 = B_ * NFR_ * 128 * HW_;
  k_relu<<<eb(n128), 256, 0, stream>>>(t128, n128);
  conv<128, 1, 1>(stream, t128, p_d2, outF, nF12, 2 * FRAME_, s1);

  // auxiliary loss scalar = 0
  k_aux<<<1, 1, 0, stream>>>(outF + (long)nSeq);
}
